// GAT_1700807050149
// MI455X (gfx1250) — matrix operand order repaired and checked by reading, not compiled, not run on hardware
//
#include <hip/hip_runtime.h>
#include <cstdint>
#include <cstddef>

// ---------------------------------------------------------------------------
// GAT forward for MI455X (gfx1250, wave32, WMMA).
// All heavy math runs through V_WMMA_F32_16X16X32_F16 (f16 in, f32 accum).
// ---------------------------------------------------------------------------

typedef __attribute__((ext_vector_type(16))) _Float16 v16h;
typedef __attribute__((ext_vector_type(8)))  _Float16 v8h;
typedef __attribute__((ext_vector_type(8)))  float    v8f;

#define NB     2048          // batch
#define NNODE  62            // real nodes
#define NPAD   64            // padded nodes (WMMA-tileable)
#define FIN    128
#define HID    512
#define NHEADS 8
#define FOUT   64
#define MROWS  (NB * NPAD)   // 131072 GEMM rows

// ---------------------------------------------------------------------------
// Fragment loaders (WMMA 16x16x32 f16, wave32; this file's contiguous k order, same for A and B).
// A (16x32, row-major source): lane L (0..15) holds row M=L, halfs 0..15 = K 0..15;
// lanes 16..31 hold the same rows with K 16..31.
// B (32x16): lane L holds column N=L%16; halfs 0..15 = 16 *consecutive* K
// values starting at K = (L<16 ? 0 : 16).
// ---------------------------------------------------------------------------

__device__ __forceinline__ v16h load_afrag(const _Float16* tile, int ld) {
  int lane = threadIdx.x & 31;
  int row  = lane & 15;
  int koff = lane & 16;                         // 0 or 16
  const _Float16* p = tile + (size_t)row * ld + koff;
  v8h lo = *(const v8h*)(p);
  v8h hi = *(const v8h*)(p + 8);
  v16h r;
#pragma unroll
  for (int i = 0; i < 8; ++i) { r[i] = lo[i]; r[i + 8] = hi[i]; }
  return r;
}

// B fragment from an N-major ("transposed") tile: tile[n][k], row stride ld.
__device__ __forceinline__ v16h load_bfrag_t(const _Float16* tile, int ld) {
  int lane = threadIdx.x & 31;
  int n  = lane & 15;
  int kb = lane & 16;                           // 0 or 16
  const _Float16* p = tile + (size_t)n * ld + kb;
  v8h lo = *(const v8h*)(p);
  v8h hi = *(const v8h*)(p + 8);
  v16h r;
#pragma unroll
  for (int i = 0; i < 8; ++i) { r[i] = lo[i]; r[i + 8] = hi[i]; }
  return r;
}

// ---------------------------------------------------------------------------
// Prep kernels
// ---------------------------------------------------------------------------

// BatchNorm over the node axis + pad rows to zero, f32 -> f16.
__global__ void prep_xbn(const float* __restrict__ x,
                         const float* __restrict__ gamma,
                         const float* __restrict__ beta,
                         const float* __restrict__ mean,
                         const float* __restrict__ var,
                         _Float16* __restrict__ A0) {
  int idx = blockIdx.x * blockDim.x + threadIdx.x;   // over NB*NPAD*FIN
  if (idx >= NB * NPAD * FIN) return;
  int k = idx & (FIN - 1);
  int n = (idx >> 7) & (NPAD - 1);
  int b = idx >> 13;
  float v = 0.f;
  if (n < NNODE) {
    float s = gamma[n] * rsqrtf(var[n] + 1e-5f);
    v = (x[((size_t)b * NNODE + n) * FIN + k] - mean[n]) * s + beta[n];
  }
  A0[idx] = (_Float16)v;
}

// Swizzle W_mlp [K,Nc] row-major f32 into WMMA B-fragment order (f16).
__global__ void prep_wfrag_mlp(const float* __restrict__ W,
                               _Float16* __restrict__ Wf, int K, int Nc) {
  int tid = blockIdx.x * blockDim.x + threadIdx.x;
  int lane = tid & 31;
  int frag = tid >> 5;                       // frag = kk * (Nc/16) + nt
  int numNT = Nc >> 4;
  if (frag >= (K >> 5) * numNT) return;
  int kk = frag / numNT, nt = frag % numNT;
  int n = nt * 16 + (lane & 15);
  int kbase = kk * 32 + (lane & 16);
  _Float16* out = Wf + ((size_t)frag * 32 + lane) * 16;
#pragma unroll
  for (int i = 0; i < 16; ++i)
    out[i] = (_Float16)W[(size_t)(kbase + i) * Nc + n];
}

// Swizzle one layer of W_gat [HEADS][HID][FOUT] into a fused [HID x HID]
// fragment matrix (column c = head*64 + d).
__global__ void prep_wfrag_gat(const float* __restrict__ Wg,
                               _Float16* __restrict__ Wf) {
  int tid = blockIdx.x * blockDim.x + threadIdx.x;
  int lane = tid & 31;
  int frag = tid >> 5;
  const int numNT = HID >> 4;                // 32
  if (frag >= (HID >> 5) * numNT) return;    // 16*32 frags
  int kk = frag / numNT, nt = frag % numNT;
  int n = nt * 16 + (lane & 15);
  int head = n >> 6, d = n & 63;
  int kbase = kk * 32 + (lane & 16);
  _Float16* out = Wf + ((size_t)frag * 32 + lane) * 16;
#pragma unroll
  for (int i = 0; i < 16; ++i)
    out[i] = (_Float16)Wg[((size_t)head * HID + (kbase + i)) * FOUT + d];
}

// ---------------------------------------------------------------------------
// Generic WMMA GEMM: C[M,Nc](f16) = A[M,K](f16,row-major) * Wfrag + bias.
// One wave computes a 16x128 output tile (8 accumulators).
// B fragments are preloaded in two groups of 4 (32 VGPRs each) so the loads
// of one group issue as a clause while the other group's WMMAs drain; A is
// double-buffered across K-steps. Peak ~150 VGPRs; launch_bounds(256,2)
// raises the allocator ceiling so nothing spills to scratch.
// ---------------------------------------------------------------------------
__global__ void __launch_bounds__(256, 2)
gemm_wmma(const _Float16* __restrict__ A,
          const _Float16* __restrict__ Wf,
          _Float16* __restrict__ C,
          const float* __restrict__ bias,
          int M, int K, int Nc, int zero_tail) {
  int wid  = (blockIdx.x * blockDim.x + threadIdx.x) >> 5;
  int lane = threadIdx.x & 31;
  int nWaveCols = Nc >> 7;                    // 128 cols per wave
  int tm = wid / nWaveCols;
  int tn = wid % nWaveCols;
  if (tm * 16 >= M) return;
  int numNT = Nc >> 4;

  v8f acc[8];
#pragma unroll
  for (int t = 0; t < 8; ++t)
#pragma unroll
    for (int r = 0; r < 8; ++r) acc[t][r] = 0.f;

  const _Float16* Arow = A + (size_t)tm * 16 * K;
  int ksteps = K >> 5;

  v16h a = load_afrag(Arow, K);
  for (int kk = 0; kk < ksteps; ++kk) {
    const _Float16* wp =
        Wf + (((size_t)kk * numNT + tn * 8) * 32 + lane) * 16;
    // group 0: B fragments 0..3
    v16h b0[4];
#pragma unroll
    for (int t = 0; t < 4; ++t) b0[t] = *(const v16h*)(wp + (size_t)t * 512);
    // next A fragment (overlaps with this step's WMMAs)
    v16h anext = a;
    if (kk + 1 < ksteps) anext = load_afrag(Arow + (size_t)(kk + 1) * 32, K);
    if (kk + 2 < ksteps)
      __builtin_prefetch(Arow + (size_t)(kk + 2) * 32, 0, 3);
#pragma unroll
    for (int t = 0; t < 4; ++t) {
      acc[t] = __builtin_amdgcn_wmma_f32_16x16x32_f16(
          false, a, false, b0[t], (short)0, acc[t], false, false);
    }
    // group 1: B fragments 4..7 (loads overlap group-0 WMMAs)
    v16h b1[4];
#pragma unroll
    for (int t = 0; t < 4; ++t)
      b1[t] = *(const v16h*)(wp + (size_t)(4 + t) * 512);
#pragma unroll
    for (int t = 0; t < 4; ++t) {
      acc[4 + t] = __builtin_amdgcn_wmma_f32_16x16x32_f16(
          false, a, false, b1[t], (short)0, acc[4 + t], false, false);
    }
    a = anext;
  }

  int rbase = (lane & 16) >> 1;               // 0 or 8
  int col0  = lane & 15;
#pragma unroll
  for (int t = 0; t < 8; ++t) {
#pragma unroll
    for (int r = 0; r < 8; ++r) {
      int row = tm * 16 + r + rbase;
      int col = tn * 128 + t * 16 + col0;
      float v = acc[t][r] + (bias ? bias[col] : 0.f);
      if (zero_tail && ((row & (NPAD - 1)) >= NNODE)) v = 0.f;
      C[(size_t)row * Nc + col] = (_Float16)v;
    }
  }
}

// ---------------------------------------------------------------------------
// Per-(head,batch,node) attention scalars: f1 = h.a1, f2 = h.a2
// ---------------------------------------------------------------------------
__global__ void compute_f12(const _Float16* __restrict__ H,
                            const float* __restrict__ a_l,
                            float* __restrict__ F1, float* __restrict__ F2) {
  int idx = blockIdx.x * blockDim.x + threadIdx.x;   // over NHEADS*NB*NNODE
  if (idx >= NHEADS * NB * NNODE) return;
  int n = idx % NNODE;
  int b = (idx / NNODE) % NB;
  int head = idx / (NNODE * NB);
  const _Float16* hp = H + ((size_t)b * NPAD + n) * HID + head * FOUT;
  const float* a1 = a_l + head * (2 * FOUT);
  const float* a2 = a1 + FOUT;
  float s1 = 0.f, s2 = 0.f;
#pragma unroll 4
  for (int d = 0; d < FOUT; ++d) {
    float hv = (float)hp[d];
    s1 += hv * a1[d];
    s2 += hv * a2[d];
  }
  size_t o = ((size_t)head * NB + b) * NPAD + n;
  F1[o] = s1;
  F2[o] = s2;
}

// ---------------------------------------------------------------------------
// Attention per (batch, head): e = leaky(f2_i + f1_j) masked, softmax over i,
// hp = att @ h via WMMA, ELU, scatter into the next h_x buffer.
// ---------------------------------------------------------------------------
__global__ void __launch_bounds__(256)
attn_kernel(const _Float16* __restrict__ H,
            const float* __restrict__ F1,
            const float* __restrict__ F2,
            const int* __restrict__ adj,
            _Float16* __restrict__ Hout) {
  int b = blockIdx.x;
  int head = blockIdx.y;
  int tid = threadIdx.x;

  __shared__ float    e_s[NPAD][NPAD + 1];      // f32 scores (conflict-free cols)
  __shared__ float    f1s[NPAD], f2s[NPAD], cinv[NPAD];
  __shared__ _Float16 att_s[NPAD][NPAD];        // f16 attention (WMMA A)
  __shared__ _Float16 ht_s[NPAD][NPAD + 8];     // h transposed: [d][j] (WMMA B)

  if (tid < NPAD) {
    int n = tid;
    size_t o = ((size_t)head * NB + b) * NPAD + n;
    f1s[n] = (n < NNODE) ? F1[o] : 0.f;
    f2s[n] = (n < NNODE) ? F2[o] : 0.f;
  }
  // Stage h^T: ht_s[d][j] = h[b, j, head*64 + d]  (padded rows of H are zero)
  for (int idx = tid; idx < NPAD * NPAD; idx += blockDim.x) {
    int d = idx & (NPAD - 1);
    int j = idx >> 6;
    ht_s[d][j] = H[((size_t)b * NPAD + j) * HID + head * FOUT + d];
  }
  __syncthreads();

  const int* adjb = adj + (size_t)b * NNODE * NNODE;
  for (int idx = tid; idx < NNODE * NNODE; idx += blockDim.x) {
    int i = idx / NNODE, j = idx % NNODE;
    float v = f2s[i] + f1s[j];
    v = (v > 0.f) ? v : 0.2f * v;                          // LeakyReLU(0.2)
    e_s[i][j] = (adjb[i * NNODE + j] > 0) ? v : -9e15f;
  }
  __syncthreads();

  // softmax over i (axis=2 of [H,B,i,j] in the reference)
  if (tid < NNODE) {
    int j = tid;
    float m = -3.0e38f;
    for (int i = 0; i < NNODE; ++i) m = fmaxf(m, e_s[i][j]);
    float s = 0.f;
    for (int i = 0; i < NNODE; ++i) {
      float t = __expf(e_s[i][j] - m);
      e_s[i][j] = t;
      s += t;
    }
    cinv[j] = 1.f / s;
  }
  __syncthreads();

  for (int idx = tid; idx < NPAD * NPAD; idx += blockDim.x) {
    int i = idx >> 6, j = idx & (NPAD - 1);
    att_s[i][j] = (i < NNODE && j < NNODE)
                      ? (_Float16)(e_s[i][j] * cinv[j])
                      : (_Float16)0.f;
  }
  __syncthreads();

  // hp[i,d] = sum_j att[i,j] * h[j,d] : 64x64x64 via WMMA (16 tiles, 8 waves)
  int wid = tid >> 5, lane = tid & 31;
  for (int q = wid; q < 16; q += 8) {
    int ti = q >> 2;                             // row tile
    int tt = q & 3;                              // col (d) tile
    v8f acc;
#pragma unroll
    for (int r = 0; r < 8; ++r) acc[r] = 0.f;
#pragma unroll
    for (int kk = 0; kk < 2; ++kk) {
      v16h a = load_afrag(&att_s[ti * 16][kk * 32], NPAD);
      v16h bf = load_bfrag_t(&ht_s[tt * 16][kk * 32], NPAD + 8);
      acc = __builtin_amdgcn_wmma_f32_16x16x32_f16(
          false, a, false, bf, (short)0, acc, false, false);
    }
    int rbase = (lane & 16) >> 1;
    int d = tt * 16 + (lane & 15);
#pragma unroll
    for (int r = 0; r < 8; ++r) {
      int i = ti * 16 + r + rbase;
      float v = acc[r];
      v = (v > 0.f) ? v : (__expf(v) - 1.f);     // ELU
      Hout[((size_t)b * NPAD + i) * HID + head * FOUT + d] = (_Float16)v;
    }
  }
}

// ---------------------------------------------------------------------------
// Pool over nodes, 512->3 head, log_softmax. One block per batch element.
// ---------------------------------------------------------------------------
__global__ void final_kernel(const _Float16* __restrict__ H,
                             const float* __restrict__ Wout,
                             const float* __restrict__ bout,
                             float* __restrict__ out) {
  int b = blockIdx.x;
  int tid = threadIdx.x;
  __shared__ float pooled[HID];
  __shared__ float lg[3];

  const _Float16* hb = H + (size_t)b * NPAD * HID;
  int c0 = tid * 2;
  float s0 = 0.f, s1 = 0.f;
  for (int n = 0; n < NNODE; ++n) {
    s0 += (float)hb[(size_t)n * HID + c0];
    s1 += (float)hb[(size_t)n * HID + c0 + 1];
  }
  pooled[c0] = s0;
  pooled[c0 + 1] = s1;
  __syncthreads();

  int wid = tid >> 5, lane = tid & 31;
  if (wid < 3) {
    float acc = 0.f;
    for (int j = lane; j < HID; j += 32) acc += pooled[j] * Wout[j * 3 + wid];
#pragma unroll
    for (int off = 16; off; off >>= 1) acc += __shfl_xor(acc, off, 32);
    if (lane == 0) lg[wid] = acc + bout[wid];
  }
  __syncthreads();
  if (tid == 0) {
    float m = fmaxf(lg[0], fmaxf(lg[1], lg[2]));
    float s = __expf(lg[0] - m) + __expf(lg[1] - m) + __expf(lg[2] - m);
    float ls = logf(s);
    out[b * 3 + 0] = lg[0] - m - ls;
    out[b * 3 + 1] = lg[1] - m - ls;
    out[b * 3 + 2] = lg[2] - m - ls;
  }
}

// ---------------------------------------------------------------------------
// Host orchestration
// ---------------------------------------------------------------------------
extern "C" void kernel_launch(void* const* d_in, const int* in_sizes, int n_in,
                              void* d_out, int out_size, void* d_ws,
                              size_t ws_size, hipStream_t stream) {
  const float* x     = (const float*)d_in[0];
  const int*   adj   = (const int*)d_in[1];
  const float* gamma = (const float*)d_in[2];
  const float* beta  = (const float*)d_in[3];
  const float* mean  = (const float*)d_in[4];
  const float* var   = (const float*)d_in[5];
  const float* Wmlp  = (const float*)d_in[6];
  const float* bmlp  = (const float*)d_in[7];
  const float* Wgat  = (const float*)d_in[8];
  const float* agat  = (const float*)d_in[9];
  const float* Wout  = (const float*)d_in[10];
  const float* bout  = (const float*)d_in[11];
  float* out = (float*)d_out;

  // Workspace carve-out (all 256B aligned)
  char* ws = (char*)d_ws;
  size_t off = 0;
  auto carve = [&](size_t bytes) -> void* {
    void* p = ws + off;
    off += (bytes + 255) & ~(size_t)255;
    return p;
  };
  _Float16* WfM = (_Float16*)carve((size_t)FIN * HID * 2);
  _Float16* WfG = (_Float16*)carve((size_t)3 * HID * HID * 2);
  _Float16* A0  = (_Float16*)carve((size_t)MROWS * FIN * 2);
  _Float16* H0  = (_Float16*)carve((size_t)MROWS * HID * 2);
  _Float16* H1  = (_Float16*)carve((size_t)MROWS * HID * 2);
  float*    F1  = (float*)carve((size_t)NHEADS * NB * NPAD * 4);
  float*    F2  = (float*)carve((size_t)NHEADS * NB * NPAD * 4);
  if (off > ws_size) return;  // insufficient scratch: bail deterministically

  // 1) Weight swizzles (tiny)
  {
    int frags = (FIN / 32) * (HID / 16);       // 128
    prep_wfrag_mlp<<<(frags * 32 + 255) / 256, 256, 0, stream>>>(Wmlp, WfM,
                                                                 FIN, HID);
  }
  for (int l = 0; l < 3; ++l) {
    int frags = (HID / 32) * (HID / 16);       // 512
    prep_wfrag_gat<<<(frags * 32 + 255) / 256, 256, 0, stream>>>(
        Wgat + (size_t)l * NHEADS * HID * FOUT, WfG + (size_t)l * HID * HID);
  }

  // 2) BN + pad -> f16
  {
    int total = NB * NPAD * FIN;
    prep_xbn<<<total / 256, 256, 0, stream>>>(x, gamma, beta, mean, var, A0);
  }

  // 3) Input MLP: H0 = A0 @ Wmlp + bmlp
  {
    int waves = (MROWS / 16) * (HID / 128);    // 32768
    gemm_wmma<<<waves * 32 / 256, 256, 0, stream>>>(A0, WfM, H0, bmlp, MROWS,
                                                    FIN, HID, 1);
  }

  // 4) GAT layers
  for (int l = 0; l < 3; ++l) {
    int waves = (MROWS / 16) * (HID / 128);
    gemm_wmma<<<waves * 32 / 256, 256, 0, stream>>>(
        H0, WfG + (size_t)l * HID * HID, H1, nullptr, MROWS, HID, HID, 1);

    int total = NHEADS * NB * NNODE;
    compute_f12<<<(total + 255) / 256, 256, 0, stream>>>(
        H1, agat + (size_t)l * NHEADS * 2 * FOUT, F1, F2);

    dim3 grid(NB, NHEADS);
    attn_kernel<<<grid, 256, 0, stream>>>(H1, F1, F2, adj, H0);
  }

  // 5) Pool + classifier head + log_softmax
  final_kernel<<<NB, 256, 0, stream>>>(H0, Wout, bout, out);
}
